// CausalSelfAttention_3968549782185
// MI455X (gfx1250) — matrix operand order repaired and checked by reading, not compiled, not run on hardware
//
#include <hip/hip_runtime.h>
#include <hip/hip_bf16.h>

typedef __attribute__((ext_vector_type(16))) __bf16 v16bf;
typedef __attribute__((ext_vector_type(8)))  float  v8f;
typedef unsigned short u16;

// ---- problem constants (fixed by reference) ----
#define BB   2
#define TT   2048
#define DD   1024
#define HH   16
#define HD   64
#define MM   (BB * TT)     // 4096 rows
#define N1   (3 * DD)      // 3072
#define KK   DD            // 1024

union Frag {
    int4  i[2];
    v16bf v;
};

static __device__ __forceinline__ u16 f32_to_bf16(float f) {
    union { float f; unsigned int u; } c;
    c.f = f;
    unsigned int u = c.u;
    u += 0x7FFFu + ((u >> 16) & 1u);   // round-to-nearest-even
    return (u16)(u >> 16);
}

static __device__ __forceinline__ float redmax16(float v) {
    v = fmaxf(v, __shfl_xor(v, 1));
    v = fmaxf(v, __shfl_xor(v, 2));
    v = fmaxf(v, __shfl_xor(v, 4));
    v = fmaxf(v, __shfl_xor(v, 8));
    return v;
}
static __device__ __forceinline__ float redsum16(float v) {
    v += __shfl_xor(v, 1);
    v += __shfl_xor(v, 2);
    v += __shfl_xor(v, 4);
    v += __shfl_xor(v, 8);
    return v;
}

static __device__ __forceinline__ v8f wmma_bf16(const v16bf& a, const v16bf& b, const v8f& c) {
    return __builtin_amdgcn_wmma_f32_16x16x32_bf16(
        false, a, false, b, (short)0, c, false, false);
}

// ---------------- f32 -> bf16 conversion ----------------
__global__ __launch_bounds__(256) void cvt_bf16_kernel(const float* __restrict__ in,
                                                       u16* __restrict__ out, int n) {
    int i = blockIdx.x * 256 + threadIdx.x;
    if (i < n) out[i] = f32_to_bf16(in[i]);
}

// =====================================================================
// Register-blocked GEMM core: each wave computes a 64(M) x 32(N) block
// as 4x2 WMMA tiles, reusing A fragments x2 and B fragments x4.
// A [*, KK] row-major bf16 ; Bw [*, KK] row-major bf16 (C = A @ Bw^T).
// =====================================================================
#define GEMM_WAVE_BODY(arow0, arow1, arow2, arow3, brow0, brow1, ACC)        \
    do {                                                                     \
        _Pragma("unroll 2")                                                  \
        for (int kk = 0; kk < KK; kk += 32) {                                \
            Frag af0, af1, af2, af3, bf0, bf1;                               \
            af0.i[0] = *(const int4*)(arow0 + kk + 8 * half);                \
            af0.i[1] = *(const int4*)(arow0 + kk + 16 + 8 * half);           \
            af1.i[0] = *(const int4*)(arow1 + kk + 8 * half);                \
            af1.i[1] = *(const int4*)(arow1 + kk + 16 + 8 * half);           \
            af2.i[0] = *(const int4*)(arow2 + kk + 8 * half);                \
            af2.i[1] = *(const int4*)(arow2 + kk + 16 + 8 * half);           \
            af3.i[0] = *(const int4*)(arow3 + kk + 8 * half);                \
            af3.i[1] = *(const int4*)(arow3 + kk + 16 + 8 * half);           \
            bf0.i[0] = *(const int4*)(brow0 + kk + 8 * half);                \
            bf0.i[1] = *(const int4*)(brow0 + kk + 16 + 8 * half);           \
            bf1.i[0] = *(const int4*)(brow1 + kk + 8 * half);                \
            bf1.i[1] = *(const int4*)(brow1 + kk + 16 + 8 * half);           \
            ACC[0][0] = wmma_bf16(af0.v, bf0.v, ACC[0][0]);                  \
            ACC[0][1] = wmma_bf16(af0.v, bf1.v, ACC[0][1]);                  \
            ACC[1][0] = wmma_bf16(af1.v, bf0.v, ACC[1][0]);                  \
            ACC[1][1] = wmma_bf16(af1.v, bf1.v, ACC[1][1]);                  \
            ACC[2][0] = wmma_bf16(af2.v, bf0.v, ACC[2][0]);                  \
            ACC[2][1] = wmma_bf16(af2.v, bf1.v, ACC[2][1]);                  \
            ACC[3][0] = wmma_bf16(af3.v, bf0.v, ACC[3][0]);                  \
            ACC[3][1] = wmma_bf16(af3.v, bf1.v, ACC[3][1]);                  \
        }                                                                    \
    } while (0)

// ---------------- GEMM1: qkv = x @ Wqkv^T, scatter to Qh/Kh/Vt ----------------
// Block: 256 threads = 8 waves arranged 4(M) x 2(N) -> 256(M) x 64(N) per block.
// Qh,Kh: [B*H, T, 64] bf16 ; Vt: [B*H, 64, T] bf16.
__global__ __launch_bounds__(256) void gemm_qkv_kernel(const u16* __restrict__ A,
                                                       const u16* __restrict__ Bw,
                                                       u16* __restrict__ Qh,
                                                       u16* __restrict__ Kh,
                                                       u16* __restrict__ Vt) {
    const int lane = threadIdx.x & 31;
    const int wave = threadIdx.x >> 5;
    const int half = lane >> 4;
    const int n16  = lane & 15;
    const int wm   = wave >> 1;     // 0..3
    const int wn   = wave & 1;      // 0..1

    const int m0 = blockIdx.x * 256 + wm * 64;
    const int n0 = blockIdx.y * 64 + wn * 32;

    const u16* arow0 = A + (size_t)(m0 + 0 * 16 + n16) * KK;
    const u16* arow1 = A + (size_t)(m0 + 1 * 16 + n16) * KK;
    const u16* arow2 = A + (size_t)(m0 + 2 * 16 + n16) * KK;
    const u16* arow3 = A + (size_t)(m0 + 3 * 16 + n16) * KK;
    const u16* brow0 = Bw + (size_t)(n0 + 0 * 16 + n16) * KK;
    const u16* brow1 = Bw + (size_t)(n0 + 1 * 16 + n16) * KK;

    v8f acc[4][2];
#pragma unroll
    for (int i = 0; i < 4; ++i)
#pragma unroll
        for (int j = 0; j < 2; ++j) acc[i][j] = (v8f){};

    GEMM_WAVE_BODY(arow0, arow1, arow2, arow3, brow0, brow1, acc);

    // scatter-store: lane holds col n, rows m = r + 8*half within each tile
#pragma unroll
    for (int j = 0; j < 2; ++j) {
        const int n   = n0 + j * 16 + n16;
        const int sec = n >> 10;        // 0=q,1=k,2=v (uniform per wave)
        const int rem = n & 1023;
        const int h   = rem >> 6;       // head (uniform per wave)
        const int d   = rem & 63;
#pragma unroll
        for (int i = 0; i < 4; ++i) {
#pragma unroll
            for (int r = 0; r < 8; ++r) {
                int gm = m0 + i * 16 + r + 8 * half;
                int b_ = gm >> 11;
                int t  = gm & (TT - 1);
                u16 val = f32_to_bf16(acc[i][j][r]);
                int bh = b_ * HH + h;
                if (sec == 0)      Qh[((size_t)bh * TT + t) * HD + d] = val;
                else if (sec == 1) Kh[((size_t)bh * TT + t) * HD + d] = val;
                else               Vt[((size_t)bh * HD + d) * TT + t] = val;
            }
        }
    }
}

// ---------------- Flash attention ----------------
// Block: 128 threads = 4 independent waves. Each wave owns TWO 16-row q-tiles
// (32 q rows) so every K/V fragment load is reused by 2 WMMAs: 16 b128 loads
// per 32-kv block feed 16 WMMAs. V fragments are held in registers across both
// q-tiles; the wave-local LDS P slice is reused sequentially (LDS ops are
// in-order within a wave; only the store->load RAW needs s_wait_dscnt).
__global__ __launch_bounds__(128) void attn_kernel(const u16* __restrict__ Qh,
                                                   const u16* __restrict__ Kh,
                                                   const u16* __restrict__ Vt,
                                                   u16* __restrict__ attn_bf) {
    const int lane = threadIdx.x & 31;
    const int wave = threadIdx.x >> 5;      // 0..3
    const int half = lane >> 4;
    const int n16  = lane & 15;

    const int q0 = blockIdx.x * 128 + wave * 32;   // 32 q rows per wave
    const int bh = blockIdx.y;                     // 0..31
    const int b  = bh >> 4;
    const int h  = bh & 15;

    const u16* Qbase = Qh + (size_t)bh * TT * HD;
    const u16* Kbase = Kh + (size_t)bh * TT * HD;
    const u16* Vbase = Vt + (size_t)bh * HD * TT;

    // Q A-fragments: [q-tile][d-step]
    Frag aq[2][2];
#pragma unroll
    for (int qt = 0; qt < 2; ++qt) {
        const u16* qr = Qbase + (size_t)(q0 + qt * 16 + n16) * HD;
#pragma unroll
        for (int ds = 0; ds < 2; ++ds) {
            aq[qt][ds].i[0] = *(const int4*)(qr + ds * 32 + 8 * half);
            aq[qt][ds].i[1] = *(const int4*)(qr + ds * 32 + 16 + 8 * half);
        }
    }

    v8f o[2][4];
#pragma unroll
    for (int qt = 0; qt < 2; ++qt)
#pragma unroll
        for (int dt = 0; dt < 4; ++dt) o[qt][dt] = (v8f){};
    float rowmax[2][8], rowsum[2][8];
#pragma unroll
    for (int qt = 0; qt < 2; ++qt)
#pragma unroll
        for (int r = 0; r < 8; ++r) { rowmax[qt][r] = -1e30f; rowsum[qt][r] = 0.0f; }

    __shared__ u16 ldsP[4][16 * 40];        // per-wave 16x32 bf16 P tile, pitch 40
    u16* myP = ldsP[wave];

    // softmax in exp2 domain: v_exp_f32 is natively 2^x
    const float scale2 = 0.125f * 1.44269504088896340736f;  // (1/sqrt(64)) * log2(e)

    for (int kv0 = 0; kv0 < q0 + 32; kv0 += 32) {
        // ---- S = Q @ K^T : K fragments shared by both q-tiles ----
        v8f sc[2][2];                       // [q-tile][kv half]
#pragma unroll
        for (int s = 0; s < 2; ++s) {
            const u16* kr = Kbase + (size_t)(kv0 + s * 16 + n16) * HD;
            Frag bk0, bk1;
            bk0.i[0] = *(const int4*)(kr + 8 * half);
            bk0.i[1] = *(const int4*)(kr + 16 + 8 * half);
            bk1.i[0] = *(const int4*)(kr + 32 + 8 * half);
            bk1.i[1] = *(const int4*)(kr + 32 + 16 + 8 * half);
#pragma unroll
            for (int qt = 0; qt < 2; ++qt) {
                v8f t = {};
                t = wmma_bf16(aq[qt][0].v, bk0.v, t);
                t = wmma_bf16(aq[qt][1].v, bk1.v, t);
                sc[qt][s] = t;
            }
        }

        // ---- V fragments: loaded once, reused by both q-tiles ----
        Frag bv[4];
#pragma unroll
        for (int dt = 0; dt < 4; ++dt) {
            const u16* vr = Vbase + (size_t)(dt * 16 + n16) * TT + kv0 + 8 * half;
            bv[dt].i[0] = *(const int4*)(vr);
            bv[dt].i[1] = *(const int4*)(vr + 16);
        }

        // ---- per q-tile: mask/scale, online softmax, P via LDS, O += P@V ----
#pragma unroll
        for (int qt = 0; qt < 2; ++qt) {
#pragma unroll
            for (int r = 0; r < 8; ++r) {
                const int qg = q0 + qt * 16 + r + 8 * half;
                float v0 = sc[qt][0][r];
                float v1 = sc[qt][1][r];
                v0 = ((kv0 + n16)      <= qg) ? v0 * scale2 : -1e30f;
                v1 = ((kv0 + 16 + n16) <= qg) ? v1 * scale2 : -1e30f;

                float m = redmax16(fmaxf(v0, v1));
                m = fmaxf(m, rowmax[qt][r]);

                float resc = exp2f(rowmax[qt][r] - m);
                float p0   = exp2f(v0 - m);
                float p1   = exp2f(v1 - m);
                rowsum[qt][r] = rowsum[qt][r] * resc + redsum16(p0 + p1);
                rowmax[qt][r] = m;
#pragma unroll
                for (int dt = 0; dt < 4; ++dt) o[qt][dt][r] *= resc;

                const int mrow = r + 8 * half;
                myP[mrow * 40 + n16]      = f32_to_bf16(p0);
                myP[mrow * 40 + 16 + n16] = f32_to_bf16(p1);
            }

            // wave-local LDS RAW: wait for DS stores before fragment read.
            // (In-order LDS => next q-tile's stores cannot pass these loads.)
            asm volatile("s_wait_dscnt 0x0" ::: "memory");

            Frag pa;
            {
                const u16* pr = &myP[n16 * 40];
                pa.i[0] = *(const int4*)(pr + 8 * half);
                pa.i[1] = *(const int4*)(pr + 16 + 8 * half);
            }

#pragma unroll
            for (int dt = 0; dt < 4; ++dt)
                o[qt][dt] = wmma_bf16(pa.v, bv[dt].v, o[qt][dt]);
        }
    }

    // ---- normalize and store bf16 [B,T,D] ----
#pragma unroll
    for (int qt = 0; qt < 2; ++qt) {
#pragma unroll
        for (int r = 0; r < 8; ++r) {
            const float inv = 1.0f / rowsum[qt][r];
            const int t = q0 + qt * 16 + r + 8 * half;
            const size_t base = ((size_t)(b * TT + t)) * DD + h * HD;
#pragma unroll
            for (int dt = 0; dt < 4; ++dt)
                attn_bf[base + dt * 16 + n16] = f32_to_bf16(o[qt][dt][r] * inv);
        }
    }
}

// ---------------- GEMM2: out = attn @ Wproj^T (f32 output) ----------------
__global__ __launch_bounds__(256) void gemm_proj_kernel(const u16* __restrict__ A,
                                                        const u16* __restrict__ Bw,
                                                        float* __restrict__ out) {
    const int lane = threadIdx.x & 31;
    const int wave = threadIdx.x >> 5;
    const int half = lane >> 4;
    const int n16  = lane & 15;
    const int wm   = wave >> 1;
    const int wn   = wave & 1;

    const int m0 = blockIdx.x * 256 + wm * 64;
    const int n0 = blockIdx.y * 64 + wn * 32;

    const u16* arow0 = A + (size_t)(m0 + 0 * 16 + n16) * KK;
    const u16* arow1 = A + (size_t)(m0 + 1 * 16 + n16) * KK;
    const u16* arow2 = A + (size_t)(m0 + 2 * 16 + n16) * KK;
    const u16* arow3 = A + (size_t)(m0 + 3 * 16 + n16) * KK;
    const u16* brow0 = Bw + (size_t)(n0 + 0 * 16 + n16) * KK;
    const u16* brow1 = Bw + (size_t)(n0 + 1 * 16 + n16) * KK;

    v8f acc[4][2];
#pragma unroll
    for (int i = 0; i < 4; ++i)
#pragma unroll
        for (int j = 0; j < 2; ++j) acc[i][j] = (v8f){};

    GEMM_WAVE_BODY(arow0, arow1, arow2, arow3, brow0, brow1, acc);

#pragma unroll
    for (int j = 0; j < 2; ++j) {
        const int n = n0 + j * 16 + n16;
#pragma unroll
        for (int i = 0; i < 4; ++i) {
#pragma unroll
            for (int r = 0; r < 8; ++r) {
                int gm = m0 + i * 16 + r + 8 * half;
                out[(size_t)gm * DD + n] = acc[i][j][r];
            }
        }
    }
}

// ---------------- host-side launch ----------------
extern "C" void kernel_launch(void* const* d_in, const int* in_sizes, int n_in,
                              void* d_out, int out_size, void* d_ws, size_t ws_size,
                              hipStream_t stream) {
    const float* x     = (const float*)d_in[0];   // [B,T,D]
    const float* Wqkv  = (const float*)d_in[1];   // [3D,D]
    const float* Wproj = (const float*)d_in[2];   // [D,D]
    float* out = (float*)d_out;

    char* ws = (char*)d_ws;
    size_t off = 0;
    auto carve = [&](size_t bytes) -> char* {
        char* p = ws + off;
        off += (bytes + 255) & ~(size_t)255;
        return p;
    };

    u16* x_bf     = (u16*)carve((size_t)MM * KK * 2);           //  8 MB
    u16* wqkv_bf  = (u16*)carve((size_t)N1 * KK * 2);           //  6 MB
    u16* wproj_bf = (u16*)carve((size_t)DD * KK * 2);           //  2 MB
    u16* Qh       = (u16*)carve((size_t)BB * HH * TT * HD * 2); //  8 MB
    u16* Kh       = (u16*)carve((size_t)BB * HH * TT * HD * 2); //  8 MB
    u16* Vt       = (u16*)carve((size_t)BB * HH * HD * TT * 2); //  8 MB
    u16* attn_bf  = (u16*)carve((size_t)MM * DD * 2);           //  8 MB

    // 1) convert f32 inputs to bf16
    {
        int n = MM * KK;
        cvt_bf16_kernel<<<(n + 255) / 256, 256, 0, stream>>>(x, x_bf, n);
        n = N1 * KK;
        cvt_bf16_kernel<<<(n + 255) / 256, 256, 0, stream>>>(Wqkv, wqkv_bf, n);
        n = DD * KK;
        cvt_bf16_kernel<<<(n + 255) / 256, 256, 0, stream>>>(Wproj, wproj_bf, n);
    }

    // 2) qkv projection -> head-major Q/K and transposed V  (256x64 per block)
    gemm_qkv_kernel<<<dim3(MM / 256, N1 / 64), 256, 0, stream>>>(x_bf, wqkv_bf, Qh, Kh, Vt);

    // 3) causal flash attention (2 q-tiles per wave, 4 waves per block)
    attn_kernel<<<dim3(TT / 128, BB * HH), 128, 0, stream>>>(Qh, Kh, Vt, attn_bf);

    // 4) output projection (f32 out)
    gemm_proj_kernel<<<dim3(MM / 256, DD / 64), 256, 0, stream>>>(attn_bf, wproj_bf, out);

    (void)in_sizes; (void)n_in; (void)out_size; (void)ws_size;
}
